// DiscretizedManifoldTransformer_33139967656632
// MI455X (gfx1250) — matrix operand order repaired and checked by reading, not compiled, not run on hardware
//
#include <hip/hip_runtime.h>

// ---------------- problem constants ----------------
constexpr int Bq = 2, Tq = 2048, Cq = 768, Hq = 12, Sq = 256, Dq = 64, Nq = 8;
constexpr int FFq = 3072, VLq = 4, NCq = 1024, NLAYERS = 4;
constexpr int Mq = Bq * Tq;                 // 4096 rows
constexpr size_t MCq = (size_t)Mq * Cq;     // 3,145,728
constexpr int CARD_ROWS = Bq * Hq * Nq * Sq; // 49152
#define EPSV 1e-5f

typedef __bf16 bf16_t;
typedef __attribute__((ext_vector_type(16))) __bf16 v16bf;
typedef __attribute__((ext_vector_type(4)))  __bf16 v4bf;
typedef __attribute__((ext_vector_type(8)))  float  v8f;

// ---------------- GEMM config ----------------
#define BMT 128
#define BNT 128
#define BKT 64
#define LSTR 72   // padded LDS stride (bf16 elems)

enum { EP_NONE = 0, EP_SIGMOID = 1, EP_FEGE = 2 };

// =====================================================================
// bf16-WMMA GEMM: Cout[M,N] = ep(A[M,K] @ B + bias) (+ addsrc)
// B row-major K x N (transB=0) or N x K (transB=1). f32 in/out.
// Requires M%128==0, N%128==0, K%64==0 (all call sites guarantee this).
// =====================================================================
__global__ __launch_bounds__(256)
void gemm_bf16_wmma(const float* __restrict__ A, int lda,
                    const float* __restrict__ Bm, int ldb, int transB,
                    const float* __restrict__ bias,
                    const float* __restrict__ addsrc,
                    float* __restrict__ Cout, int ldc,
                    int M, int N, int K, int ep,
                    const float* __restrict__ alphaPtr)
{
    __shared__ bf16_t sA[BMT * LSTR];   // [m][k]
    __shared__ bf16_t sB[BNT * LSTR];   // transposed: [n][k]

    const int tid  = threadIdx.x;
    const int lane = tid & 31;
    const int wave = tid >> 5;          // 0..7
    const int wm   = wave >> 1;         // 0..3  (rows of 32)
    const int wn   = wave & 1;          // 0..1  (cols of 64)
    const int l15  = lane & 15;
    const int hsel = lane >> 4;

    const int mBase = blockIdx.y * BMT;
    const int nBase = blockIdx.x * BNT;

    const float alpha = (ep == EP_FEGE && alphaPtr) ? alphaPtr[0] : 0.f;

    v8f acc[2][4];
    const v8f vzero = {0.f,0.f,0.f,0.f,0.f,0.f,0.f,0.f};
#pragma unroll
    for (int mt = 0; mt < 2; ++mt)
#pragma unroll
        for (int nt = 0; nt < 4; ++nt) acc[mt][nt] = vzero;

    for (int k0 = 0; k0 < K; k0 += BKT) {
        // ---- A: 128x64, vectorized float4 -> v4bf ----
        float4 va[8];
#pragma unroll
        for (int i = 0; i < 8; ++i) {
            int chunk = i * 256 + tid;
            int k4 = (chunk & 15) * 4, m = chunk >> 4;
            va[i] = *(const float4*)(A + (size_t)(mBase + m) * lda + k0 + k4);
        }
#pragma unroll
        for (int i = 0; i < 8; ++i) {
            int chunk = i * 256 + tid;
            int k4 = (chunk & 15) * 4, m = chunk >> 4;
            v4bf pk;
            pk[0] = (bf16_t)va[i].x; pk[1] = (bf16_t)va[i].y;
            pk[2] = (bf16_t)va[i].z; pk[3] = (bf16_t)va[i].w;
            *(v4bf*)&sA[m * LSTR + k4] = pk;
        }
        if (transB) {
            float4 vb[8];
#pragma unroll
            for (int i = 0; i < 8; ++i) {
                int chunk = i * 256 + tid;
                int k4 = (chunk & 15) * 4, n = chunk >> 4;
                vb[i] = *(const float4*)(Bm + (size_t)(nBase + n) * ldb + k0 + k4);
            }
#pragma unroll
            for (int i = 0; i < 8; ++i) {
                int chunk = i * 256 + tid;
                int k4 = (chunk & 15) * 4, n = chunk >> 4;
                v4bf pk;
                pk[0] = (bf16_t)vb[i].x; pk[1] = (bf16_t)vb[i].y;
                pk[2] = (bf16_t)vb[i].z; pk[3] = (bf16_t)vb[i].w;
                *(v4bf*)&sB[n * LSTR + k4] = pk;
            }
        } else {
            float4 vb[8];
#pragma unroll
            for (int i = 0; i < 8; ++i) {
                int chunk = i * 256 + tid;
                int n4 = (chunk & 31) * 4, kk = chunk >> 5;
                vb[i] = *(const float4*)(Bm + (size_t)(k0 + kk) * ldb + nBase + n4);
            }
#pragma unroll
            for (int i = 0; i < 8; ++i) {
                int chunk = i * 256 + tid;
                int n4 = (chunk & 31) * 4, kk = chunk >> 5;
                sB[(n4 + 0) * LSTR + kk] = (bf16_t)vb[i].x;
                sB[(n4 + 1) * LSTR + kk] = (bf16_t)vb[i].y;
                sB[(n4 + 2) * LSTR + kk] = (bf16_t)vb[i].z;
                sB[(n4 + 3) * LSTR + kk] = (bf16_t)vb[i].w;
            }
        }

        // ---- prefetch next K-tile into cache while this stage computes ----
        int kn = k0 + BKT;
        if (kn < K) {
            int pm = tid >> 1, pk2 = (tid & 1) * 32;
            __builtin_prefetch(A + (size_t)(mBase + pm) * lda + kn + pk2, 0, 3);
            if (transB) {
                __builtin_prefetch(Bm + (size_t)(nBase + pm) * ldb + kn + pk2, 0, 3);
            } else {
                int pk3 = tid >> 2, pn = (tid & 3) * 32;
                __builtin_prefetch(Bm + (size_t)(kn + pk3) * ldb + nBase + pn, 0, 3);
            }
        }
        __syncthreads();

        // ---- 2 k-subtiles of 32, 8 WMMAs each ----
#pragma unroll
        for (int ks = 0; ks < 2; ++ks) {
            v16bf afrag[2], bfrag[4];
#pragma unroll
            for (int mt = 0; mt < 2; ++mt) {
                int m  = wm * 32 + mt * 16 + l15;
                int kb = ks * 32 + hsel * 16;
                const bf16_t* pa = &sA[m * LSTR + kb];
#pragma unroll
                for (int j = 0; j < 16; ++j) {
                    afrag[mt][j]     = pa[j];
                }
            }
#pragma unroll
            for (int nt = 0; nt < 4; ++nt) {
                int n  = wn * 64 + nt * 16 + l15;
                int kb = ks * 32 + hsel * 16;
                const bf16_t* pb = &sB[n * LSTR + kb];
#pragma unroll
                for (int j = 0; j < 16; ++j) bfrag[nt][j] = pb[j];
            }
#pragma unroll
            for (int mt = 0; mt < 2; ++mt)
#pragma unroll
                for (int nt = 0; nt < 4; ++nt)
                    acc[mt][nt] = __builtin_amdgcn_wmma_f32_16x16x32_bf16(
                        false, afrag[mt], false, bfrag[nt],
                        (short)0, acc[mt][nt], false, false);
        }
        __syncthreads();
    }

    // ---- epilogue + store (row = v + 8*hsel, col = l15) ----
#pragma unroll
    for (int mt = 0; mt < 2; ++mt) {
#pragma unroll
        for (int nt = 0; nt < 4; ++nt) {
            int gn = nBase + wn * 64 + nt * 16 + l15;
            float bv = bias ? bias[gn] : 0.f;
#pragma unroll
            for (int v = 0; v < 8; ++v) {
                int gm = mBase + wm * 32 + mt * 16 + v + 8 * hsel;
                float y = acc[mt][nt][v] + bv;
                if (ep == EP_SIGMOID)      y = 1.f / (1.f + __expf(-y));
                else if (ep == EP_FEGE)    y = y + alpha * y * __expf(-0.5f * y * y);
                if (addsrc) y += addsrc[(size_t)gm * ldc + gn];
                Cout[(size_t)gm * ldc + gn] = y;
            }
        }
    }
}

// ---------------- helpers ----------------
__device__ __forceinline__ float blockSum256(float v, float* red) {
    int t = threadIdx.x;
    red[t] = v; __syncthreads();
#pragma unroll
    for (int off = 128; off > 0; off >>= 1) {
        if (t < off) red[t] += red[t + off];
        __syncthreads();
    }
    float r = red[0]; __syncthreads();
    return r;
}
__device__ __forceinline__ float blockSum64(float v, float* red) {
    int t = threadIdx.x;
    red[t] = v; __syncthreads();
#pragma unroll
    for (int off = 32; off > 0; off >>= 1) {
        if (t < off) red[t] += red[t + off];
        __syncthreads();
    }
    float r = red[0]; __syncthreads();
    return r;
}

// row LayerNorm over C, optional residual add: out = addres + ln(in)*g+b
__global__ __launch_bounds__(256)
void ln_kernel(const float* __restrict__ in, const float* __restrict__ g,
               const float* __restrict__ b, const float* __restrict__ addres,
               float* __restrict__ out, int C)
{
    __shared__ float red[256];
    size_t row = blockIdx.x;
    const float* x = in + row * C;
    int t = threadIdx.x;
    float s = 0.f;
    for (int c = t; c < C; c += 256) s += x[c];
    float mean = blockSum256(s, red) / (float)C;
    float v = 0.f;
    for (int c = t; c < C; c += 256) { float d = x[c] - mean; v += d * d; }
    float var = blockSum256(v, red) / (float)C;
    float rs = rsqrtf(var + EPSV);
    for (int c = t; c < C; c += 256) {
        float y = (x[c] - mean) * rs * g[c] + b[c];
        if (addres) y += addres[row * C + c];
        out[row * C + c] = y;
    }
}

// per-(b,h,n,d) cumulative sum of gates*marks over s; emit lc and chunk sums
__global__ __launch_bounds__(64)
void chunk_scan_kernel(const float* __restrict__ marks, const float* __restrict__ gates,
                       float* __restrict__ lc, float* __restrict__ chunks)
{
    int n = blockIdx.x % Nq;
    int h = (blockIdx.x / Nq) % Hq;
    int b = blockIdx.x / (Nq * Hq);
    int d = threadIdx.x;
    size_t baseT = (size_t)b * Tq + (size_t)n * Sq;
    size_t outB  = (size_t)blockIdx.x * Sq * Dq;
    float acc = 0.f;
    for (int s = 0; s < Sq; ++s) {
        size_t gi = (baseT + s) * Cq + h * Dq + d;
        acc += gates[gi] * marks[gi];
        lc[outB + (size_t)s * Dq + d] = acc;
    }
    chunks[(size_t)blockIdx.x * Dq + d] = acc;
}

// exclusive prefix over chunks (n) + LayerNorm over D -> ncar
__global__ __launch_bounds__(64)
void carry_ln_kernel(const float* __restrict__ chunks, const float* __restrict__ g,
                     const float* __restrict__ b, float* __restrict__ ncar)
{
    __shared__ float red[64];
    int bh = blockIdx.x, d = threadIdx.x;
    float acc = 0.f;
    for (int n = 0; n < Nq; ++n) {
        float carry = acc;
        acc += chunks[((size_t)bh * Nq + n) * Dq + d];
        float mean = blockSum64(carry, red) / (float)Dq;
        float diff = carry - mean;
        float var  = blockSum64(diff * diff, red) / (float)Dq;
        ncar[((size_t)bh * Nq + n) * Dq + d] = diff * rsqrtf(var + EPSV) * g[d] + b[d];
    }
}

// cards = LN_D(ncar + shift(lc)); comb = [xh | cards]
__global__ __launch_bounds__(64)
void comb_kernel(const float* __restrict__ xn, const float* __restrict__ lc,
                 const float* __restrict__ ncar, const float* __restrict__ cg,
                 const float* __restrict__ cb, float* __restrict__ comb)
{
    __shared__ float red[64];
    size_t row = blockIdx.x;
    int s = (int)(row % Sq); size_t bhn = row / Sq;
    int n = (int)(bhn % Nq); size_t bh = bhn / Nq;
    int h = (int)(bh % Hq);  int b = (int)(bh / Hq);
    int d = threadIdx.x;
    float val = ncar[bhn * Dq + d] +
                (s > 0 ? lc[bhn * (size_t)Sq * Dq + (size_t)(s - 1) * Dq + d] : 0.f);
    float mean = blockSum64(val, red) / (float)Dq;
    float diff = val - mean;
    float var  = blockSum64(diff * diff, red) / (float)Dq;
    float normed = diff * rsqrtf(var + EPSV) * cg[d] + cb[d];
    size_t crow = row * (size_t)(2 * Dq);
    comb[crow + Dq + d] = normed;
    comb[crow + d] = xn[((size_t)b * Tq + (size_t)n * Sq + s) * Cq + h * Dq + d];
}

// (B,H,N,S,D) -> (B,T,C); ho has leading dim 128 (padded h2 output)
__global__ __launch_bounds__(64)
void card_out_kernel(const float* __restrict__ ho, float* __restrict__ out)
{
    size_t row = blockIdx.x;
    int s = (int)(row % Sq); size_t bhn = row / Sq;
    int n = (int)(bhn % Nq); size_t bh = bhn / Nq;
    int h = (int)(bh % Hq);  int b = (int)(bh / Hq);
    int d = threadIdx.x;
    out[((size_t)b * Tq + (size_t)n * Sq + s) * Cq + h * Dq + d] = ho[row * 128 + d];
}

// pad h2W (128x64) -> (128x128), h2b (64) -> (128), zeros beyond Dq
__global__ __launch_bounds__(256)
void pad_h2_kernel(const float* __restrict__ w, const float* __restrict__ b,
                   float* __restrict__ wp, float* __restrict__ bp)
{
    int i = blockIdx.x * 256 + threadIdx.x;   // 0 .. 128*128-1
    int k = i >> 7, n = i & 127;
    wp[i] = (n < Dq) ? w[k * Dq + n] : 0.f;
    if (i < 128) bp[i] = (i < Dq) ? b[i] : 0.f;
}

__global__ void add_kernel(const float* __restrict__ a, const float* __restrict__ b,
                           float* __restrict__ o, size_t n)
{
    size_t i = (size_t)blockIdx.x * blockDim.x + threadIdx.x;
    if (i < n) o[i] = a[i] + b[i];
}

__global__ __launch_bounds__(256)
void colsq_kernel(const float* __restrict__ E, float* __restrict__ csq)
{
    __shared__ float red[256];
    int j = blockIdx.x;
    float s = 0.f;
    for (int c = threadIdx.x; c < Cq; c += 256) {
        float v = E[(size_t)j * Cq + c]; s += v * v;
    }
    float t = blockSum256(s, red);
    if (threadIdx.x == 0) csq[j] = t;
}

__global__ __launch_bounds__(256)
void argmin_kernel(const float* __restrict__ dots, const float* __restrict__ csq,
                   int* __restrict__ idx)
{
    __shared__ float bv[256]; __shared__ int bi[256];
    int m = blockIdx.x, t = threadIdx.x;
    float best = 3.4e38f; int bidx = 0;
    for (int j = t; j < NCq; j += 256) {
        float d = csq[j] - 2.f * dots[(size_t)m * NCq + j];
        if (d < best) { best = d; bidx = j; }
    }
    bv[t] = best; bi[t] = bidx; __syncthreads();
#pragma unroll
    for (int off = 128; off > 0; off >>= 1) {
        if (t < off) {
            if (bv[t + off] < bv[t] || (bv[t + off] == bv[t] && bi[t + off] < bi[t])) {
                bv[t] = bv[t + off]; bi[t] = bi[t + off];
            }
        }
        __syncthreads();
    }
    if (t == 0) idx[m] = bi[0];
}

// total (+)= q ; res -= q ; partial[m] = sum (q-res)^2
__global__ __launch_bounds__(256)
void vq_update_kernel(const float* __restrict__ E, const int* __restrict__ idx,
                      float* __restrict__ res, float* __restrict__ total,
                      int initTotal, float* __restrict__ partial)
{
    __shared__ float red[256];
    int m = blockIdx.x;
    const float* e = E + (size_t)idx[m] * Cq;
    float ss = 0.f;
    for (int c = threadIdx.x; c < Cq; c += 256) {
        size_t i = (size_t)m * Cq + c;
        float q = e[c];
        float diff = q - res[i];
        ss += diff * diff;
        if (initTotal) total[i] = q; else total[i] += q;
        res[i] -= q;
    }
    float t = blockSum256(ss, red);
    if (threadIdx.x == 0) partial[m] = t;
}

__global__ __launch_bounds__(256)
void reduce_add_kernel(const float* __restrict__ partial, int n, float scale,
                       float* __restrict__ qlel)
{
    __shared__ float red[256];
    float s = 0.f;
    for (int i = threadIdx.x; i < n; i += 256) s += partial[i];
    float t = blockSum256(s, red);
    if (threadIdx.x == 0) { qlel[0] += t * scale; qlel[1] += t * scale; }
}

__global__ void init2_kernel(float* p) { p[0] = 0.f; p[1] = 0.f; }

// ---------------- host-side helpers ----------------
static inline void launch_gemm(const float* A, int lda, const float* Bm, int ldb,
                               int transB, const float* bias, const float* addsrc,
                               float* C, int ldc, int M, int N, int K, int ep,
                               const float* alphaPtr, hipStream_t s)
{
    dim3 g(N / BNT, M / BMT);
    gemm_bf16_wmma<<<g, 256, 0, s>>>(A, lda, Bm, ldb, transB, bias, addsrc,
                                     C, ldc, M, N, K, ep, alphaPtr);
}

struct CardParams {
    const float *mW, *mb, *gW, *gb, *carry_g, *carry_b, *card_g, *card_b;
    const float *h1W, *h1b, *ha, *h2Wp, *h2bp, *pW, *pb, *lnW_g, *lnW_b;
};

static void run_card(const float* xn, float* aout, float* s1, float* s2,
                     float* big, float* chunks, float* ncar,
                     const CardParams& p, hipStream_t st)
{
    float* marks = s1;
    float* gates = s2;
    float* lc    = big + 2 * MCq;          // dead before hbuf is written
    float* comb  = big;                    // 49152 x 128 = 2*MC
    float* hbuf  = big + 2 * MCq;          // 49152 x 128
    float* ho    = big;                    // 49152 x 128 (overwrites dead comb)
    float* ctmp  = s2;                     // gates dead after scan
    float* pout  = big + 2 * MCq;          // hbuf dead after h2 gemm

    launch_gemm(xn, Cq, p.mW, Cq, 0, p.mb, nullptr, marks, Cq, Mq, Cq, Cq, EP_NONE, nullptr, st);
    launch_gemm(xn, Cq, p.gW, Cq, 0, p.gb, nullptr, gates, Cq, Mq, Cq, Cq, EP_SIGMOID, nullptr, st);
    chunk_scan_kernel<<<Bq * Hq * Nq, 64, 0, st>>>(marks, gates, lc, chunks);
    carry_ln_kernel<<<Bq * Hq, 64, 0, st>>>(chunks, p.carry_g, p.carry_b, ncar);
    comb_kernel<<<CARD_ROWS, 64, 0, st>>>(xn, lc, ncar, p.card_g, p.card_b, comb);
    launch_gemm(comb, 2 * Dq, p.h1W, 2 * Dq, 0, p.h1b, nullptr, hbuf, 2 * Dq,
                CARD_ROWS, 2 * Dq, 2 * Dq, EP_FEGE, p.ha, st);
    // h2 with N padded 64 -> 128 (zero columns), output ldc = 128
    launch_gemm(hbuf, 2 * Dq, p.h2Wp, 128, 0, p.h2bp, nullptr, ho, 128,
                CARD_ROWS, 128, 2 * Dq, EP_NONE, nullptr, st);
    card_out_kernel<<<CARD_ROWS, 64, 0, st>>>(ho, ctmp);
    launch_gemm(ctmp, Cq, p.pW, Cq, 0, p.pb, nullptr, pout, Cq, Mq, Cq, Cq, EP_NONE, nullptr, st);
    // aout = xn + LN(pout)
    ln_kernel<<<Mq, 256, 0, st>>>(pout, p.lnW_g, p.lnW_b, xn, aout, Cq);
}

extern "C" void kernel_launch(void* const* d_in, const int* in_sizes, int n_in,
                              void* d_out, int out_size, void* d_ws, size_t ws_size,
                              hipStream_t stream)
{
    const float* xin = (const float*)d_in[0];
    auto P = [&](int l, int k) -> const float* {
        return (const float*)d_in[1 + l * 29 + k];
    };

    float* w   = (float*)d_ws;
    float* XA  = w; w += MCq;
    float* XB  = w; w += MCq;
    float* P1  = w; w += MCq;
    float* P2  = w; w += MCq;
    float* P3  = w; w += MCq;
    float* P4  = w; w += MCq;
    float* BIG = w; w += 4 * MCq;          // M x FF / comb+hbuf+ho / dots
    float* chunks  = w; w += (size_t)Bq * Hq * Nq * Dq;
    float* ncar    = w; w += (size_t)Bq * Hq * Nq * Dq;
    float* csq     = w; w += NCq;
    float* partial = w; w += Mq;
    float* h2Wp    = w; w += 128 * 128;
    float* h2bp    = w; w += 128;
    int*   idxb    = (int*)w;

    float* outx = (float*)d_out;
    float* qlel = outx + MCq;              // [ql, el]

    init2_kernel<<<1, 1, 0, stream>>>(qlel);
    hipMemcpyAsync(XA, xin, MCq * sizeof(float), hipMemcpyDeviceToDevice, stream);

    float* x  = XA;
    float* xo = XB;
    const size_t addBlocks = (MCq + 255) / 256;
    const float vqScale = 1.f / ((float)VLq * (float)Mq * (float)Cq);

    for (int l = 0; l < NLAYERS; ++l) {
        CardParams cp;
        cp.mW = P(l,0);  cp.mb = P(l,1);  cp.gW = P(l,2);  cp.gb = P(l,3);
        cp.carry_g = P(l,4); cp.carry_b = P(l,5); cp.card_g = P(l,6); cp.card_b = P(l,7);
        cp.h1W = P(l,8); cp.h1b = P(l,9); cp.ha = P(l,10);
        cp.h2Wp = h2Wp; cp.h2bp = h2bp;
        cp.pW = P(l,13); cp.pb = P(l,14);
        cp.lnW_g = P(l,15); cp.lnW_b = P(l,16);
        const float *ln1_g = P(l,17), *ln1_b = P(l,18);
        const float *ln2_g = P(l,19), *ln2_b = P(l,20);
        const float *ln3_g = P(l,21), *ln3_b = P(l,22);
        const float *m1W = P(l,23), *m1b = P(l,24), *ma = P(l,25);
        const float *m2W = P(l,26), *m2b = P(l,27), *books = P(l,28);

        // pad this layer's h2 weights to a full 128-wide tile
        pad_h2_kernel<<<64, 256, 0, stream>>>(P(l,11), P(l,12), h2Wp, h2bp);

        // a = card(ln1(x))
        ln_kernel<<<Mq, 256, 0, stream>>>(x, ln1_g, ln1_b, nullptr, P1, Cq);
        run_card(P1, P2, P3, P4, BIG, chunks, ncar, cp, stream);
        // a = card(ln1(x + a)) ; x = x + a
        add_kernel<<<addBlocks, 256, 0, stream>>>(x, P2, P1, MCq);
        ln_kernel<<<Mq, 256, 0, stream>>>(P1, ln1_g, ln1_b, nullptr, P2, Cq);
        run_card(P2, P1, P3, P4, BIG, chunks, ncar, cp, stream);
        add_kernel<<<addBlocks, 256, 0, stream>>>(x, P1, xo, MCq);
        { float* t = x; x = xo; xo = t; }

        // MLP: x = x + fege(ln2(x) @ m1W + m1b, ma) @ m2W + m2b
        ln_kernel<<<Mq, 256, 0, stream>>>(x, ln2_g, ln2_b, nullptr, P1, Cq);
        launch_gemm(P1, Cq, m1W, FFq, 0, m1b, nullptr, BIG, FFq,
                    Mq, FFq, Cq, EP_FEGE, ma, stream);
        launch_gemm(BIG, FFq, m2W, Cq, 0, m2b, x, xo, Cq,
                    Mq, Cq, FFq, EP_NONE, nullptr, stream);
        { float* t = x; x = xo; xo = t; }

        // RVQ on ln3(x); res in P1, total in P2, dots in BIG
        ln_kernel<<<Mq, 256, 0, stream>>>(x, ln3_g, ln3_b, nullptr, P1, Cq);
        for (int i = 0; i < VLq; ++i) {
            const float* E = books + (size_t)i * NCq * Cq;
            colsq_kernel<<<NCq, 256, 0, stream>>>(E, csq);
            launch_gemm(P1, Cq, E, Cq, 1, nullptr, nullptr, BIG, NCq,
                        Mq, NCq, Cq, EP_NONE, nullptr, stream);
            argmin_kernel<<<Mq, 256, 0, stream>>>(BIG, csq, idxb);
            vq_update_kernel<<<Mq, 256, 0, stream>>>(E, idxb, P1, P2, (i == 0) ? 1 : 0, partial);
            reduce_add_kernel<<<1, 256, 0, stream>>>(partial, Mq, vqScale, qlel);
        }
        // x = x + total   (ste forward value == total)
        add_kernel<<<addBlocks, 256, 0, stream>>>(x, P2, xo, MCq);
        { float* t = x; x = xo; xo = t; }
    }

    hipMemcpyAsync(outx, x, MCq * sizeof(float), hipMemcpyDeviceToDevice, stream);
}
